// CircuitTransformer_24404004176271
// MI455X (gfx1250) — matrix operand order repaired and checked by reading, not compiled, not run on hardware
//
#include <hip/hip_runtime.h>
#include <stdint.h>

// ---------------- constants ----------------
#define D_MODEL 768
#define N_HEADS 12
#define D_HEAD  64
#define D_MLP   3072
#define PREV    14
#define NSTRM   26          // PREV + N_HEADS
#define OUTSTRM 27
#define BATCH   4
#define SEQ     1024
#define NTOK    (BATCH*SEQ) // 4096
#define LN_EPS  1e-5f
#define NEG_BIG (-1.0e30f)

typedef __bf16 bf16;
typedef __attribute__((ext_vector_type(16))) __bf16 v16bf;
typedef __attribute__((ext_vector_type(8)))  float  v8f;

#define DEV static __device__ __forceinline__

DEV v8f wmma_bf16(v16bf a, v16bf b, v8f c) {
  return __builtin_amdgcn_wmma_f32_16x16x32_bf16(false, a, false, b, (short)0, c, false, false);
}

union frag_u { v16bf v; uint4 q[2]; };
union pack8_u { bf16 h8[8]; uint4 q; };

// A fragment: 16(M) x 32(K), bf16 row-major, row stride lda (elements).
// lane<16 holds row=lane, K {0..7,16..23}; lane>=16 same row, K {8..15,24..31}.
// -> two contiguous 16B loads per lane.
DEV v16bf load_a_frag(const bf16* A, int lda) {
  const int lane = threadIdx.x & 31;
  const int row  = lane & 15;
  const int kh   = (lane >> 4) << 3;       // 0 or 8
  const bf16* r  = A + (long)row * lda + kh;
  frag_u u;
  u.q[0] = *(const uint4*)(r);
  u.q[1] = *(const uint4*)(r + 16);
  return u.v;
}

// B fragment: 32(K) x 16(N) where source is K-contiguous per column:
// element(k,col) = S[col*ld + k]. Per lane: K {0..7,16..23} or {8..15,24..31}.
DEV v16bf load_bT_frag(const bf16* S, int ld) {
  const int lane = threadIdx.x & 31;
  const int col  = lane & 15;
  const int k0   = (lane >> 4) << 3;       // 0 or 8
  const bf16* r  = S + (long)col * ld + k0;
  frag_u u;
  u.q[0] = *(const uint4*)(r);
  u.q[1] = *(const uint4*)(r + 16);
  return u.v;
}

// ---------------- kernel: f32 -> bf16 batched transpose -------------------
// src: [B][K][N] f32  ->  dst: [B][N][K] bf16
__global__ void k_cvt_T(const float* __restrict__ src, bf16* __restrict__ dst,
                        int Bc, int K, int N) {
  long i = (long)blockIdx.x * blockDim.x + threadIdx.x;
  const long total = (long)Bc * K * N;
  if (i >= total) return;
  const long kn = (long)K * N;
  const int  b  = (int)(i / kn);
  const long r  = i % kn;
  const int  k  = (int)(r / N);
  const int  n  = (int)(r % N);
  dst[((long)b * N + n) * K + k] = (bf16)src[i];
}

// ---------------- kernel: copy resid -> out streams 0..13 ----------------
__global__ void k_copy_resid(const float* __restrict__ resid, float* __restrict__ out) {
  const long per_tok = PREV * D_MODEL / 4;          // 2688 float4
  long i = (long)blockIdx.x * blockDim.x + threadIdx.x;
  if (i >= (long)NTOK * per_tok) return;
  long t = i / per_tok, j = i % per_tok;
  ((float4*)out)[t * (OUTSTRM*D_MODEL/4) + j] = ((const float4*)resid)[i];
}

// ---------------- kernel: masked stream mix + LayerNorm (36 combos) -------
// one block per token; one wave per combo (8 waves round-robin over 36)
__global__ __launch_bounds__(256) void k_streams_ln(
    const float* __restrict__ resid,
    const float* __restrict__ mq, const float* __restrict__ mk,
    const float* __restrict__ mv, bf16* __restrict__ nbuf) {
  __shared__ __align__(16) float lds[PREV * D_MODEL];
  const int t = blockIdx.x, tid = threadIdx.x;
  const float* rt = resid + (long)t * (PREV * D_MODEL);
  for (int i = tid; i < PREV * D_MODEL; i += 256) lds[i] = rt[i];
  __syncthreads();
  const int w = tid >> 5, lane = tid & 31;
  const float* mlog[3] = { mq, mk, mv };
  for (int c = w; c < 36; c += 8) {
    const int mat = c / 12, h = c % 12;
    const float* lg = mlog[mat];
    float msk[PREV];
#pragma unroll
    for (int n = 0; n < PREV; ++n) msk[n] = (lg[n*N_HEADS + h] > 0.f) ? 1.f : 0.f;
    float x[24], s = 0.f, s2 = 0.f;
#pragma unroll
    for (int i = 0; i < 24; ++i) {
      const int d = lane + i*32;
      float acc = 0.f;
#pragma unroll
      for (int n = 0; n < PREV; ++n) acc += msk[n] * lds[n*D_MODEL + d];
      x[i] = acc; s += acc; s2 += acc*acc;
    }
#pragma unroll
    for (int o = 16; o; o >>= 1) { s += __shfl_xor(s, o, 32); s2 += __shfl_xor(s2, o, 32); }
    const float mean = s * (1.f/D_MODEL);
    const float rstd = rsqrtf(s2 * (1.f/D_MODEL) - mean*mean + LN_EPS);
    bf16* orow = nbuf + ((long)(mat*N_HEADS + h) * NTOK + t) * D_MODEL;
#pragma unroll
    for (int i = 0; i < 24; ++i) orow[lane + i*32] = (bf16)((x[i]-mean)*rstd);
  }
}

// Ping-pong pipelined 16xN GEMM body over K (KSTEPS must be even).
// A: row-major [16 x K], Wn: N-major [64(n) x K]. acc[4] = 16x64 tile.
DEV void gemm_16x64_pingpong(const bf16* Arow, const bf16* Wn, int ld, int ksteps,
                             v8f acc[4]) {
  v16bf a0 = load_a_frag(Arow, ld), a1;
  v16bf b0[4], b1[4];
#pragma unroll
  for (int nt = 0; nt < 4; ++nt) b0[nt] = load_bT_frag(Wn + (long)(nt*16)*ld, ld);
  for (int kk = 0; kk < ksteps; kk += 2) {
    const int k1 = (kk+1 < ksteps) ? (kk+1)*32 : 0;
    a1 = load_a_frag(Arow + k1, ld);
#pragma unroll
    for (int nt = 0; nt < 4; ++nt) b1[nt] = load_bT_frag(Wn + (long)(nt*16)*ld + k1, ld);
#pragma unroll
    for (int nt = 0; nt < 4; ++nt) acc[nt] = wmma_bf16(a0, b0[nt], acc[nt]);
    const int k2 = (kk+2 < ksteps) ? (kk+2)*32 : 0;
    a0 = load_a_frag(Arow + k2, ld);
#pragma unroll
    for (int nt = 0; nt < 4; ++nt) b0[nt] = load_bT_frag(Wn + (long)(nt*16)*ld + k2, ld);
#pragma unroll
    for (int nt = 0; nt < 4; ++nt) acc[nt] = wmma_bf16(a1, b1[nt], acc[nt]);
  }
}

// ---------------- kernel: QKV projection + bias + rotary ------------------
// weights pre-transposed: Wt [head][64(e)][768(k)]
// q,k stored [b][h][s][e]; v stored transposed [b][h][e][s]
__global__ __launch_bounds__(256) void k_qkv(
    const bf16* __restrict__ nbuf,
    const bf16* __restrict__ wqt, const bf16* __restrict__ wkt, const bf16* __restrict__ wvt,
    const float* __restrict__ bq, const float* __restrict__ bk, const float* __restrict__ bv,
    bf16* __restrict__ qb, bf16* __restrict__ kb, bf16* __restrict__ vbt) {
  int bid = blockIdx.x;                               // 3*12*32
  const int mat = bid / (N_HEADS*32); bid %= N_HEADS*32;
  const int h = bid / 32, tt = bid % 32;
  const int w = threadIdx.x >> 5;
  const int t0 = tt*128 + w*16;
  const bf16* Wt   = (mat==0 ? wqt : mat==1 ? wkt : wvt) + (long)h * D_HEAD * D_MODEL;
  const float* bia = (mat==0 ? bq : mat==1 ? bk : bv) + h * D_HEAD;
  const bf16* Arow = nbuf + ((long)(mat*N_HEADS + h) * NTOK + t0) * D_MODEL;
  v8f acc[4] = {};
  gemm_16x64_pingpong(Arow, Wt, D_MODEL, D_MODEL/32, acc);
  const int lane = threadIdx.x & 31, col = lane & 15;
  float res[4][8];
#pragma unroll
  for (int nt = 0; nt < 4; ++nt)
#pragma unroll
    for (int v = 0; v < 8; ++v)
      res[nt][v] = acc[nt][v] + bia[nt*16 + col];
  if (mat < 2) {                                      // rotary on q,k
    float rot[4][8];
#pragma unroll
    for (int nt = 0; nt < 4; ++nt) {
      const int e = nt*16 + col, j = e & 31;
      const float inv = __expf(-(float)j * 0.28782313662425576f); // 10000^(-j/32)
#pragma unroll
      for (int v = 0; v < 8; ++v) {
        const int r = ((lane>>4)<<3) + v;
        const int s = (t0 + r) & (SEQ-1);
        const float ang = (float)s * inv;
        const float sn = __sinf(ang), cs = __cosf(ang);
        const float flip = (nt < 2) ? -res[nt+2][v] : res[nt-2][v];
        rot[nt][v] = res[nt][v]*cs + flip*sn;
      }
    }
    bf16* O = (mat==0 ? qb : kb);
#pragma unroll
    for (int nt = 0; nt < 4; ++nt)
#pragma unroll
      for (int v = 0; v < 8; ++v) {
        const int r = ((lane>>4)<<3) + v;
        const int t = t0 + r, b = t >> 10, s = t & (SEQ-1);
        O[(((long)b*N_HEADS + h)*SEQ + s)*D_HEAD + nt*16 + col] = (bf16)rot[nt][v];
      }
  } else {                                            // V: transposed + vectorized store
    const int b  = t0 >> 10;
    const int s0 = (t0 & (SEQ-1)) + ((lane>>4)<<3);   // 8 consecutive positions
#pragma unroll
    for (int nt = 0; nt < 4; ++nt) {
      pack8_u pv;
#pragma unroll
      for (int v = 0; v < 8; ++v) pv.h8[v] = (bf16)res[nt][v];
      const int e = nt*16 + col;
      *(uint4*)(vbt + (((long)b*N_HEADS + h)*D_HEAD + e)*SEQ + s0) = pv.q;
    }
  }
}

// ---------------- kernel: flash attention per (b,h) -----------------------
__global__ __launch_bounds__(256) void k_attn(
    const bf16* __restrict__ qb, const bf16* __restrict__ kb,
    const bf16* __restrict__ vbt, bf16* __restrict__ zb) {
  __shared__ __align__(16) bf16 lp[8][16*32];
  int bid = blockIdx.x;                               // 4*12*8
  const int b = bid / (N_HEADS*8); int rem = bid % (N_HEADS*8);
  const int h = rem / 8, qg = rem % 8;
  const int w = threadIdx.x >> 5;
  const int q0 = (qg*8 + w) * 16;
  const long base = (((long)b*N_HEADS + h) * SEQ) * D_HEAD;
  const bf16* Q  = qb  + base;
  const bf16* K  = kb  + base;
  const bf16* Vt = vbt + base;                        // [e][s]
  const v16bf aq0 = load_a_frag(Q + (long)q0*D_HEAD, D_HEAD);
  const v16bf aq1 = load_a_frag(Q + (long)q0*D_HEAD + 32, D_HEAD);
  float m[8], l[8];
#pragma unroll
  for (int v = 0; v < 8; ++v) { m[v] = NEG_BIG; l[v] = 0.f; }
  v8f accz[4] = {};
  const int lane = threadIdx.x & 31, col = lane & 15;
  const int nkt = (q0 + 15)/32 + 1;
  for (int kc = 0; kc < nkt; ++kc) {
    // prefetch V fragments for this chunk (independent of scores/softmax)
    v16bf bv[4];
#pragma unroll
    for (int nt = 0; nt < 4; ++nt)
      bv[nt] = load_bT_frag(Vt + (long)(nt*16)*SEQ + kc*32, SEQ);
    float p[2][8], tmax[8];
#pragma unroll
    for (int v = 0; v < 8; ++v) tmax[v] = NEG_BIG;
#pragma unroll
    for (int hf = 0; hf < 2; ++hf) {
      const int kbase = kc*32 + hf*16;
      v8f s = {};
      s = wmma_bf16(aq0, load_bT_frag(K + (long)kbase*D_HEAD,      D_HEAD), s);
      s = wmma_bf16(aq1, load_bT_frag(K + (long)kbase*D_HEAD + 32, D_HEAD), s);
#pragma unroll
      for (int v = 0; v < 8; ++v) {
        float sv = s[v] * 0.125f;                      // 1/sqrt(64)
        const int rtok = q0 + ((lane>>4)<<3) + v;
        const int ctok = kbase + col;
        if (ctok > rtok) sv = NEG_BIG;
        p[hf][v] = sv;
        tmax[v] = fmaxf(tmax[v], sv);
      }
    }
#pragma unroll
    for (int v = 0; v < 8; ++v) {
#pragma unroll
      for (int o = 1; o < 16; o <<= 1) tmax[v] = fmaxf(tmax[v], __shfl_xor(tmax[v], o, 32));
      const float mn = fmaxf(m[v], tmax[v]);
      const float e0 = __expf(p[0][v] - mn);
      const float e1 = __expf(p[1][v] - mn);
      float rs = e0 + e1;
#pragma unroll
      for (int o = 1; o < 16; o <<= 1) rs += __shfl_xor(rs, o, 32);
      const float alpha = __expf(m[v] - mn);
      l[v] = l[v]*alpha + rs; m[v] = mn;
#pragma unroll
      for (int nt = 0; nt < 4; ++nt) accz[nt][v] *= alpha;
      const int r = ((lane>>4)<<3) + v;
      lp[w][r*32 + col]      = (bf16)e0;               // P tile (16x32), wave-private
      lp[w][r*32 + col + 16] = (bf16)e1;
    }
    const v16bf ap = load_a_frag(&lp[w][0], 32);
#pragma unroll
    for (int nt = 0; nt < 4; ++nt)
      accz[nt] = wmma_bf16(ap, bv[nt], accz[nt]);
  }
#pragma unroll
  for (int nt = 0; nt < 4; ++nt)
#pragma unroll
    for (int v = 0; v < 8; ++v) {
      const int r = ((lane>>4)<<3) + v;
      const long t = (long)b*SEQ + q0 + r;
      zb[((long)h*NTOK + t)*D_HEAD + nt*16 + col] = (bf16)(accz[nt][v] / l[v]);
    }
}

// ---------------- kernel: z @ W_O -> out streams 14..25 -------------------
// W_O pre-transposed: wot [head][768(m)][64(e)]
__global__ __launch_bounds__(256) void k_attnout(
    const bf16* __restrict__ zb, const bf16* __restrict__ wot,
    const float* __restrict__ bo, float* __restrict__ out) {
  int bid = blockIdx.x;                                // 12*32
  const int h = bid / 32, tt = bid % 32;
  const int w = threadIdx.x >> 5;
  const int t0 = tt*128 + w*16;
  const v16bf a0 = load_a_frag(zb + ((long)h*NTOK + t0)*D_HEAD,      D_HEAD);
  const v16bf a1 = load_a_frag(zb + ((long)h*NTOK + t0)*D_HEAD + 32, D_HEAD);
  const bf16* Wt = wot + (long)h * D_MODEL * D_HEAD;
  const int lane = threadIdx.x & 31, col = lane & 15;
  // ping-pong pipelined over N tiles (48, even)
  v16bf b00 = load_bT_frag(Wt,      D_HEAD);
  v16bf b01 = load_bT_frag(Wt + 32, D_HEAD);
  v16bf b10, b11;
  for (int nt = 0; nt < D_MODEL/16; nt += 2) {
    const long o1 = (long)((nt+1)*16)*D_HEAD;
    b10 = load_bT_frag(Wt + o1,      D_HEAD);
    b11 = load_bT_frag(Wt + o1 + 32, D_HEAD);
    v8f acc = {};
    acc = wmma_bf16(a0, b00, acc);
    acc = wmma_bf16(a1, b01, acc);
#pragma unroll
    for (int v = 0; v < 8; ++v) {
      const int r = ((lane>>4)<<3) + v;
      const int e = nt*16 + col;
      out[((long)(t0+r)*OUTSTRM + PREV + h)*D_MODEL + e] = acc[v] + bo[e]*(1.f/N_HEADS);
    }
    const long o2 = (nt+2 < D_MODEL/16) ? (long)((nt+2)*16)*D_HEAD : 0;
    b00 = load_bT_frag(Wt + o2,      D_HEAD);
    b01 = load_bT_frag(Wt + o2 + 32, D_HEAD);
    v8f acc2 = {};
    acc2 = wmma_bf16(a0, b10, acc2);
    acc2 = wmma_bf16(a1, b11, acc2);
#pragma unroll
    for (int v = 0; v < 8; ++v) {
      const int r = ((lane>>4)<<3) + v;
      const int e = (nt+1)*16 + col;
      out[((long)(t0+r)*OUTSTRM + PREV + h)*D_MODEL + e] = acc2[v] + bo[e]*(1.f/N_HEADS);
    }
  }
}

// ---------------- kernel: MLP masked stream sum + LayerNorm ---------------
// one wave per token (block = 8 tokens)
__global__ __launch_bounds__(256) void k_mlpin_ln(
    const float* __restrict__ out, const float* __restrict__ mm,
    bf16* __restrict__ hln) {
  const int w = threadIdx.x >> 5, lane = threadIdx.x & 31;
  const long t = (long)blockIdx.x*8 + w;
  float msk[NSTRM];
#pragma unroll
  for (int n = 0; n < NSTRM; ++n) msk[n] = (mm[n] > 0.f) ? 1.f : 0.f;
  float x[24], s = 0.f, s2 = 0.f;
  for (int i = 0; i < 24; ++i) {
    const int d = lane + i*32;
    float acc = 0.f;
#pragma unroll
    for (int n = 0; n < NSTRM; ++n) acc += msk[n] * out[(t*OUTSTRM + n)*D_MODEL + d];
    x[i] = acc; s += acc; s2 += acc*acc;
  }
#pragma unroll
  for (int o = 16; o; o >>= 1) { s += __shfl_xor(s, o, 32); s2 += __shfl_xor(s2, o, 32); }
  const float mean = s * (1.f/D_MODEL);
  const float rstd = rsqrtf(s2 * (1.f/D_MODEL) - mean*mean + LN_EPS);
#pragma unroll
  for (int i = 0; i < 24; ++i)
    hln[t*D_MODEL + lane + i*32] = (bf16)((x[i]-mean)*rstd);
}

// ---------------- kernel: MLP GEMM1 + gelu --------------------------------
// W_in pre-transposed: wit [3072(n)][768(k)]
__global__ __launch_bounds__(256) void k_mlp1(
    const bf16* __restrict__ hln, const bf16* __restrict__ wit,
    const float* __restrict__ bin, bf16* __restrict__ gel) {
  int bid = blockIdx.x;                                // 32 * 48
  const int tt = bid / (D_MLP/64), ng = bid % (D_MLP/64);
  const int w = threadIdx.x >> 5;
  const int t0 = tt*128 + w*16, n0 = ng*64;
  v8f acc[4] = {};
  gemm_16x64_pingpong(hln + (long)t0*D_MODEL, wit + (long)n0*D_MODEL,
                      D_MODEL, D_MODEL/32, acc);
  const int lane = threadIdx.x & 31, col = lane & 15;
#pragma unroll
  for (int nt = 0; nt < 4; ++nt)
#pragma unroll
    for (int v = 0; v < 8; ++v) {
      const int r = ((lane>>4)<<3) + v;
      const int e = n0 + nt*16 + col;
      const float xv = acc[nt][v] + bin[e];
      const float g = 0.5f * xv * (1.f + erff(xv * 0.70710678118654752f));
      gel[(long)(t0 + r)*D_MLP + e] = (bf16)g;
    }
}

// ---------------- kernel: MLP GEMM2 -> out stream 26 ----------------------
// W_out pre-transposed: wut [768(n)][3072(k)]
__global__ __launch_bounds__(256) void k_mlp2(
    const bf16* __restrict__ gel, const bf16* __restrict__ wut,
    const float* __restrict__ bout, float* __restrict__ out) {
  int bid = blockIdx.x;                                // 32 * 12
  const int tt = bid / (D_MODEL/64), ng = bid % (D_MODEL/64);
  const int w = threadIdx.x >> 5;
  const int t0 = tt*128 + w*16, n0 = ng*64;
  v8f acc[4] = {};
  gemm_16x64_pingpong(gel + (long)t0*D_MLP, wut + (long)n0*D_MLP,
                      D_MLP, D_MLP/32, acc);
  const int lane = threadIdx.x & 31, col = lane & 15;
#pragma unroll
  for (int nt = 0; nt < 4; ++nt)
#pragma unroll
    for (int v = 0; v < 8; ++v) {
      const int r = ((lane>>4)<<3) + v;
      const int e = n0 + nt*16 + col;
      out[((long)(t0 + r)*OUTSTRM + NSTRM)*D_MODEL + e] = acc[nt][v] + bout[e];
    }
}

// ---------------- host launcher -------------------------------------------
extern "C" void kernel_launch(void* const* d_in, const int* in_sizes, int n_in,
                              void* d_out, int out_size, void* d_ws, size_t ws_size,
                              hipStream_t stream) {
  const float* resid = (const float*)d_in[0];
  const float* W_Q   = (const float*)d_in[1];
  const float* b_Q   = (const float*)d_in[2];
  const float* W_K   = (const float*)d_in[3];
  const float* b_K   = (const float*)d_in[4];
  const float* W_V   = (const float*)d_in[5];
  const float* b_V   = (const float*)d_in[6];
  const float* W_O   = (const float*)d_in[7];
  const float* b_O   = (const float*)d_in[8];
  const float* W_in  = (const float*)d_in[9];
  const float* b_in  = (const float*)d_in[10];
  const float* W_out = (const float*)d_in[11];
  const float* b_out = (const float*)d_in[12];
  const float* mql   = (const float*)d_in[13];
  const float* mkl   = (const float*)d_in[14];
  const float* mvl   = (const float*)d_in[15];
  const float* mml   = (const float*)d_in[16];
  float* out = (float*)d_out;

  // workspace layout (bf16 buffers, all sizes 256B aligned)
  const long SZ_W    = (long)N_HEADS * D_MODEL * D_HEAD;     // 589824
  const long SZ_WIN  = (long)D_MODEL * D_MLP;                // 2359296
  const long SZ_N    = 3L * N_HEADS * NTOK * D_MODEL;        // 113246208
  const long SZ_QKV  = (long)BATCH * N_HEADS * SEQ * D_HEAD; // 3145728
  const long SZ_H    = (long)NTOK * D_MODEL;
  const long SZ_G    = (long)NTOK * D_MLP;
  uint8_t* p = (uint8_t*)d_ws;
  bf16* wqt  = (bf16*)p; p += SZ_W   * 2;
  bf16* wkt  = (bf16*)p; p += SZ_W   * 2;
  bf16* wvt  = (bf16*)p; p += SZ_W   * 2;
  bf16* wot  = (bf16*)p; p += SZ_W   * 2;
  bf16* wit  = (bf16*)p; p += SZ_WIN * 2;
  bf16* wut  = (bf16*)p; p += SZ_WIN * 2;
  bf16* nbuf = (bf16*)p; p += SZ_N   * 2;
  bf16* qb   = (bf16*)p; p += SZ_QKV * 2;
  bf16* kbuf = (bf16*)p; p += SZ_QKV * 2;
  bf16* vbt  = (bf16*)p; p += SZ_QKV * 2;
  bf16* zb   = (bf16*)p; p += SZ_QKV * 2;
  bf16* hln  = (bf16*)p; p += SZ_H   * 2;
  bf16* gel  = (bf16*)p; p += SZ_G   * 2;

  auto cvtT = [&](const float* s, bf16* d, int Bc, int K, int N) {
    long n = (long)Bc * K * N;
    k_cvt_T<<<dim3((unsigned)((n + 255)/256)), dim3(256), 0, stream>>>(s, d, Bc, K, N);
  };
  cvtT(W_Q,   wqt, N_HEADS, D_MODEL, D_HEAD);   // -> [h][64][768]
  cvtT(W_K,   wkt, N_HEADS, D_MODEL, D_HEAD);
  cvtT(W_V,   wvt, N_HEADS, D_MODEL, D_HEAD);
  cvtT(W_O,   wot, N_HEADS, D_HEAD,  D_MODEL);  // -> [h][768][64]
  cvtT(W_in,  wit, 1,       D_MODEL, D_MLP);    // -> [3072][768]
  cvtT(W_out, wut, 1,       D_MLP,   D_MODEL);  // -> [768][3072]

  {
    long n4 = (long)NTOK * PREV * D_MODEL / 4;
    k_copy_resid<<<dim3((unsigned)((n4 + 255)/256)), dim3(256), 0, stream>>>(resid, out);
  }

  k_streams_ln<<<dim3(NTOK), dim3(256), 0, stream>>>(resid, mql, mkl, mvl, nbuf);

  k_qkv<<<dim3(3*N_HEADS*32), dim3(256), 0, stream>>>(
      nbuf, wqt, wkt, wvt, b_Q, b_K, b_V, qb, kbuf, vbt);

  k_attn<<<dim3(BATCH*N_HEADS*8), dim3(256), 0, stream>>>(qb, kbuf, vbt, zb);

  k_attnout<<<dim3(N_HEADS*32), dim3(256), 0, stream>>>(zb, wot, b_O, out);

  k_mlpin_ln<<<dim3(NTOK/8), dim3(256), 0, stream>>>(out, mml, hln);

  k_mlp1<<<dim3(32*(D_MLP/64)), dim3(256), 0, stream>>>(hln, wit, b_in, gel);

  k_mlp2<<<dim3(32*(D_MODEL/64)), dim3(256), 0, stream>>>(gel, wut, b_out, out);
}
